// Templ_emb_30193620091041
// MI455X (gfx1250) — matrix operand order repaired and checked by reading, not compiled, not run on hardware
//
#include <hip/hip_runtime.h>
#include <hip/hip_bf16.h>
#include <math.h>

typedef __attribute__((ext_vector_type(16))) __bf16 v16bf;
typedef __attribute__((ext_vector_type(8)))  float  v8f;

#define LL 192
#define TT 10
#define DD 64
#define DFF 256
#define NP 8            // pairs per block
#define MROWS (NP*TT)   // 80 rows
#define MT 5            // 80/16 m-tiles
#define NTHREADS 256
#define NWAVES 8

// One 16x16 output tile, K-loop of v_wmma_f32_16x16x32_bf16.
// Arow: &A[(m_base + lane%16)*lda + 8*(lane/16)]   (row-major, K contiguous)
// Brow: &BT[(n_base + lane%16)*ldb + 16*(lane/16)] (weight stored transposed, K contiguous)
__device__ __forceinline__ v8f wmma_kloop(const __bf16* Arow, const __bf16* Brow, int K) {
  v8f acc = {};
  for (int k0 = 0; k0 < K; k0 += 32) {
    v16bf a, b;
#pragma unroll
    for (int e = 0; e < 8; ++e) { a[e] = Arow[k0 + e]; a[e + 8] = Arow[k0 + 16 + e]; }
#pragma unroll
    for (int e = 0; e < 8; ++e) { const int hb = 8 * ((threadIdx.x & 31) >> 4); b[e] = Brow[k0 - hb + e]; b[e + 8] = Brow[k0 - hb + 16 + e]; }
    acc = __builtin_amdgcn_wmma_f32_16x16x32_bf16(false, a, false, b, (short)0, acc,
                                                  false, false);
  }
  return acc;
}

// LayerNorm over D=64, one thread per row (80 rows).
__device__ __forceinline__ void ln_rows(const float* src, float* dstF, __bf16* dstB,
                                        const float* g, const float* b, int tid) {
  if (tid < MROWS) {
    const float* r = src + tid * DD;
    float m = 0.f;
    for (int d = 0; d < DD; ++d) m += r[d];
    m *= (1.f / DD);
    float v = 0.f;
    for (int d = 0; d < DD; ++d) { float t = r[d] - m; v += t * t; }
    v *= (1.f / DD);
    float inv = rsqrtf(v + 1e-5f);
    for (int d = 0; d < DD; ++d) {
      float y = (r[d] - m) * inv * g[d] + b[d];
      if (dstF) dstF[tid * DD + d] = y;
      if (dstB) dstB[tid * DD + d] = (__bf16)y;
    }
  }
}

__global__ __launch_bounds__(NTHREADS)
void templ_fused_kernel(const float* __restrict__ t1d, const float* __restrict__ t2d,
                        const int* __restrict__ idx,
                        const float* __restrict__ proj_w, const float* __restrict__ proj_b,
                        const float* __restrict__ wq, const float* __restrict__ bq,
                        const float* __restrict__ wk, const float* __restrict__ bk,
                        const float* __restrict__ awv, const float* __restrict__ abv,
                        const float* __restrict__ wo, const float* __restrict__ bo,
                        const float* __restrict__ ln1_g, const float* __restrict__ ln1_b,
                        const float* __restrict__ ln2_g, const float* __restrict__ ln2_b,
                        const float* __restrict__ ff1_w, const float* __restrict__ ff1_b,
                        const float* __restrict__ ff2_w, const float* __restrict__ ff2_b,
                        const float* __restrict__ lnf_g, const float* __restrict__ lnf_b,
                        const float* __restrict__ tov_w, const float* __restrict__ tov_b,
                        const float* __restrict__ tou_w,
                        float* __restrict__ out)
{
  __shared__ float  xres[MROWS * DD];                  // residual stream, fp32 (20 KB)
  __shared__ __bf16 xb[MROWS * DD];                    // GEMM A operand (10 KB)
  __shared__ __align__(16) float  poolf[3 * MROWS * DD]; // 60 KB union pool
  __shared__ __align__(16) __bf16 wpool[DFF * DD];     // 32 KB weight pool (transposed)
  __shared__ float vu[MROWS];
  __shared__ float alphas[MROWS];

  const int tid  = threadIdx.x;
  const int lane = tid & 31;
  const int wave = tid >> 5;
  const int l16  = lane & 15;
  const int half = lane >> 4;
  const int nbase = blockIdx.x * NP;

  float*  qS    = poolf;
  float*  kS    = poolf + MROWS * DD;
  float*  vS    = poolf + 2 * MROWS * DD;
  __bf16* featb = (__bf16*)poolf;     // 80x32 bf16 (K padded 17->32)
  __bf16* ffh   = (__bf16*)poolf;     // 80x256 bf16 (after q/k/v are dead)
  float*  vp    = poolf;              // 80x64 fp32 (after ffh is dead)

  // ---------------- Stage A: pairwise feature build + projection ----------------
  for (int w = tid; w < MROWS * 32; w += NTHREADS) {
    int row = w >> 5, c = w & 31;
    int p = row / TT, t = row % TT;
    int n = nbase + p, i = n / LL, j = n % LL;
    float val;
    if (c < 10)       val = t2d[(((size_t)t * LL + i) * LL + j) * 10 + c];
    else if (c < 13)  val = t1d[((size_t)t * LL + i) * 3 + (c - 10)];
    else if (c < 16)  val = t1d[((size_t)t * LL + j) * 3 + (c - 13)];
    else if (c == 16) val = logf(fabsf((float)(idx[i] - idx[j])) + 1.0f);
    else              val = 0.f;
    featb[row * 32 + c] = (__bf16)val;
  }
  for (int w = tid; w < DD * 32; w += NTHREADS) {     // projT[n][k] (zero-pad k>=17)
    int n = w >> 5, kk = w & 31;
    wpool[n * 32 + kk] = (__bf16)(kk < 17 ? proj_w[kk * DD + n] : 0.f);
  }
  __syncthreads();
  for (int tile = wave; tile < MT * 4; tile += NWAVES) {
    int tm = tile % MT, tn = tile / MT;
    v8f acc = wmma_kloop(featb + (tm * 16 + l16) * 32 + 8 * half,
                         wpool + (tn * 16 + l16) * 32 + 16 * half, 32);
    float bias = proj_b[tn * 16 + l16];
#pragma unroll
    for (int r = 0; r < 8; ++r)
      xres[(tm * 16 + r + 8 * half) * DD + tn * 16 + l16] = acc[r] + bias;
  }
  __syncthreads();

  // ---------------- Stage B: LN1 ----------------
  ln_rows(xres, (float*)nullptr, xb, ln1_g, ln1_b, tid);
  __syncthreads();

  // ---------------- Stage C: Q, K, V GEMMs ----------------
  {
    const float* Ws[3]  = {wq, wk, awv};
    const float* Bsp[3] = {bq, bk, abv};
    float* Os[3] = {qS, kS, vS};
    for (int s = 0; s < 3; ++s) {
      for (int w = tid; w < DD * DD; w += NTHREADS) { // wT[n][k]
        int kk = w >> 6, n = w & 63;
        wpool[n * DD + kk] = (__bf16)Ws[s][w];
      }
      __syncthreads();
      for (int tile = wave; tile < MT * 4; tile += NWAVES) {
        int tm = tile % MT, tn = tile / MT;
        v8f acc = wmma_kloop(xb + (tm * 16 + l16) * DD + 8 * half,
                             wpool + (tn * 16 + l16) * DD + 16 * half, DD);
        float bias = Bsp[s][tn * 16 + l16];
#pragma unroll
        for (int r = 0; r < 8; ++r)
          Os[s][(tm * 16 + r + 8 * half) * DD + tn * 16 + l16] = acc[r] + bias;
      }
      __syncthreads();
    }
  }

  // ---------------- Stage D: attention over T (VALU, T=10, DH=16) ----------------
  for (int w = tid; w < NP * 4 * TT; w += NTHREADS) {
    int p = w / 40, rem = w % 40, h = rem / TT, qt = rem % TT;
    int base = p * TT;
    const float* qrow = qS + (base + qt) * DD + h * 16;
    float sc[TT]; float mx = -1e30f;
    for (int kt = 0; kt < TT; ++kt) {
      const float* krow = kS + (base + kt) * DD + h * 16;
      float s = 0.f;
      for (int d = 0; d < 16; ++d) s += qrow[d] * krow[d];
      s *= 0.25f;                       // 1/sqrt(16)
      sc[kt] = s; mx = fmaxf(mx, s);
    }
    float den = 0.f;
    for (int kt = 0; kt < TT; ++kt) { sc[kt] = __expf(sc[kt] - mx); den += sc[kt]; }
    float inv = 1.f / den;
    for (int d = 0; d < 16; ++d) {
      float o = 0.f;
      for (int kt = 0; kt < TT; ++kt) o += sc[kt] * vS[(base + kt) * DD + h * 16 + d];
      xb[(base + qt) * DD + h * 16 + d] = (__bf16)(o * inv);  // xb reused as O operand
    }
  }
  __syncthreads();

  // ---------------- Stage E: O-projection + residual ----------------
  for (int w = tid; w < DD * DD; w += NTHREADS) {
    int kk = w >> 6, n = w & 63;
    wpool[n * DD + kk] = (__bf16)wo[w];
  }
  __syncthreads();
  for (int tile = wave; tile < MT * 4; tile += NWAVES) {
    int tm = tile % MT, tn = tile / MT;
    v8f acc = wmma_kloop(xb + (tm * 16 + l16) * DD + 8 * half,
                         wpool + (tn * 16 + l16) * DD + 16 * half, DD);
    float bias = bo[tn * 16 + l16];
#pragma unroll
    for (int r = 0; r < 8; ++r)
      xres[(tm * 16 + r + 8 * half) * DD + tn * 16 + l16] += acc[r] + bias;
  }
  __syncthreads();

  // ---------------- Stage F: LN2 ----------------
  ln_rows(xres, (float*)nullptr, xb, ln2_g, ln2_b, tid);
  __syncthreads();

  // ---------------- Stage G: FFN up (64->256) + ReLU ----------------
  for (int w = tid; w < DD * DFF; w += NTHREADS) {   // ff1T[n][k], n<256, k<64
    int kk = w >> 8, n = w & 255;
    wpool[n * DD + kk] = (__bf16)ff1_w[w];
  }
  __syncthreads();
  for (int tile = wave; tile < MT * 16; tile += NWAVES) {
    int tm = tile % MT, tn = tile / MT;
    v8f acc = wmma_kloop(xb + (tm * 16 + l16) * DD + 8 * half,
                         wpool + (tn * 16 + l16) * DD + 16 * half, DD);
    float bias = ff1_b[tn * 16 + l16];
#pragma unroll
    for (int r = 0; r < 8; ++r)
      ffh[(tm * 16 + r + 8 * half) * DFF + tn * 16 + l16] =
          (__bf16)fmaxf(acc[r] + bias, 0.f);
  }
  __syncthreads();

  // ---------------- Stage H: FFN down (256->64) + residual ----------------
  for (int w = tid; w < DFF * DD; w += NTHREADS) {   // ff2T[n][k], n<64, k<256
    int kk = w >> 6, n = w & 63;
    wpool[n * DFF + kk] = (__bf16)ff2_w[w];
  }
  __syncthreads();
  for (int tile = wave; tile < MT * 4; tile += NWAVES) {
    int tm = tile % MT, tn = tile / MT;
    v8f acc = wmma_kloop(ffh + (tm * 16 + l16) * DFF + 8 * half,
                         wpool + (tn * 16 + l16) * DFF + 16 * half, DFF);
    float bias = ff2_b[tn * 16 + l16];
#pragma unroll
    for (int r = 0; r < 8; ++r)
      xres[(tm * 16 + r + 8 * half) * DD + tn * 16 + l16] += acc[r] + bias;
  }
  __syncthreads();

  // ---------------- Stage I: final LN (in place + bf16 copy) ----------------
  ln_rows(xres, xres, xb, lnf_g, lnf_b, tid);
  __syncthreads();

  // ---------------- Stage J: tov GEMM + tanh ----------------
  for (int w = tid; w < DD * DD; w += NTHREADS) {
    int kk = w >> 6, n = w & 63;
    wpool[n * DD + kk] = (__bf16)tov_w[w];
  }
  __syncthreads();
  for (int tile = wave; tile < MT * 4; tile += NWAVES) {
    int tm = tile % MT, tn = tile / MT;
    v8f acc = wmma_kloop(xb + (tm * 16 + l16) * DD + 8 * half,
                         wpool + (tn * 16 + l16) * DD + 16 * half, DD);
    float bias = tov_b[tn * 16 + l16];
#pragma unroll
    for (int r = 0; r < 8; ++r)
      vp[(tm * 16 + r + 8 * half) * DD + tn * 16 + l16] = tanhf(acc[r] + bias);
  }
  __syncthreads();

  // ---------------- Stage K: pooling logits + softmax over T ----------------
  if (tid < MROWS) {
    const float* r = vp + tid * DD;
    float s = 0.f;
    for (int d = 0; d < DD; ++d) s += r[d] * tou_w[d];
    vu[tid] = s;
  }
  __syncthreads();
  if (tid < NP) {
    float mx = -1e30f;
    for (int t = 0; t < TT; ++t) mx = fmaxf(mx, vu[tid * TT + t]);
    float den = 0.f; float e[TT];
    for (int t = 0; t < TT; ++t) { e[t] = __expf(vu[tid * TT + t] - mx); den += e[t]; }
    float inv = 1.f / den;
    for (int t = 0; t < TT; ++t) alphas[tid * TT + t] = e[t] * inv;
  }
  __syncthreads();

  // ---------------- Stage L: weighted sum over T -> output ----------------
  for (int w = tid; w < NP * DD; w += NTHREADS) {
    int p = w >> 6, d = w & 63;
    float acc = 0.f;
    for (int t = 0; t < TT; ++t) acc += alphas[p * TT + t] * xres[(p * TT + t) * DD + d];
    out[(size_t)(nbase + p) * DD + d] = acc;
  }
}

extern "C" void kernel_launch(void* const* d_in, const int* in_sizes, int n_in,
                              void* d_out, int out_size, void* d_ws, size_t ws_size,
                              hipStream_t stream) {
  (void)in_sizes; (void)n_in; (void)out_size; (void)d_ws; (void)ws_size;
  const float* t1d    = (const float*)d_in[0];
  const float* t2d    = (const float*)d_in[1];
  const int*   idx    = (const int*)  d_in[2];
  const float* proj_w = (const float*)d_in[3];
  const float* proj_b = (const float*)d_in[4];
  const float* wq     = (const float*)d_in[5];
  const float* bq     = (const float*)d_in[6];
  const float* wk     = (const float*)d_in[7];
  const float* bk     = (const float*)d_in[8];
  const float* awv    = (const float*)d_in[9];
  const float* abv    = (const float*)d_in[10];
  const float* wo     = (const float*)d_in[11];
  const float* bo     = (const float*)d_in[12];
  const float* ln1_g  = (const float*)d_in[13];
  const float* ln1_b  = (const float*)d_in[14];
  const float* ln2_g  = (const float*)d_in[15];
  const float* ln2_b  = (const float*)d_in[16];
  const float* ff1_w  = (const float*)d_in[17];
  const float* ff1_b  = (const float*)d_in[18];
  const float* ff2_w  = (const float*)d_in[19];
  const float* ff2_b  = (const float*)d_in[20];
  const float* lnf_g  = (const float*)d_in[21];
  const float* lnf_b  = (const float*)d_in[22];
  const float* tov_w  = (const float*)d_in[23];
  const float* tov_b  = (const float*)d_in[24];
  const float* tou_w  = (const float*)d_in[25];

  dim3 grid((LL * LL) / NP);   // 4608 blocks, 8 pairs each
  dim3 block(NTHREADS);
  templ_fused_kernel<<<grid, block, 0, stream>>>(
      t1d, t2d, idx, proj_w, proj_b, wq, bq, wk, bk, awv, abv, wo, bo,
      ln1_g, ln1_b, ln2_g, ln2_b, ff1_w, ff1_b, ff2_w, ff2_b,
      lnf_g, lnf_b, tov_w, tov_b, tou_w, (float*)d_out);
}
